// K_graph_Multi_ATT_88837103550597
// MI455X (gfx1250) — matrix operand order repaired and checked by reading, not compiled, not run on hardware
//
#include <hip/hip_runtime.h>
#include <hip/hip_bf16.h>
#include <stdint.h>

// ---------------------------------------------------------------------------
// MI455X / gfx1250. wave32. All GEMMs via v_wmma_f32_16x16x32_bf16 with
// operands pre-packed so every WMMA input is loaded as contiguous b128s.
// ---------------------------------------------------------------------------

typedef __attribute__((ext_vector_type(16))) __bf16 v16bf;
typedef __attribute__((ext_vector_type(8)))  float  v8f;

#define DEV __device__ __forceinline__

constexpr int   Bn    = 1024;
constexpr int   NUMCc = 24;
constexpr int   CATCc = 8;
constexpr int   Hd    = 128;
constexpr int   Kk    = 16;
constexpr int   CATVv = 100;
constexpr int   Cc    = 32;           // NUMC + CATC
constexpr float EPSf  = 1e-5f;

// ---- helpers ---------------------------------------------------------------

DEV __bf16 f2bf(float f) {
  unsigned u = __float_as_uint(f);
  unsigned r = u + 0x7FFFu + ((u >> 16) & 1u);   // round-to-nearest-even
  unsigned short hs = (unsigned short)(r >> 16);
  return __builtin_bit_cast(__bf16, hs);
}

DEV float waveRedSum(float v) {
  #pragma unroll
  for (int o = 16; o; o >>= 1) v += __shfl_xor(v, o, 32);
  return v;
}
DEV float halfRedSum(float v) {   // reduce within a 16-lane half of wave32
  #pragma unroll
  for (int o = 8; o; o >>= 1) v += __shfl_xor(v, o, 32);
  return v;
}

// CDNA5 WMMA: D = A(16x32 bf16) * B(32x16 bf16) + C(16x16 f32)
DEV v8f wmma_bf16(v16bf a, v16bf b, v8f c) {
  return __builtin_amdgcn_wmma_f32_16x16x32_bf16(false, a, false, b, (short)0, c,
                                                 false, false);
}

// 16-bit A layout (ISA 7.12.2): lane L holds row m=L&15; half h maps to
//   k = (h>=8 ? 16 : 0) + (L>>4)*8 + (h&7)
// i.e. two contiguous 8-element runs -> two aligned 16B loads per lane.
DEV v16bf ld_a16(const __bf16* __restrict__ row, int k0, int lane) {
  int g8 = (lane >> 4) << 3;
  union { uint4 q[2]; v16bf v; } u;
  u.q[0] = *(const uint4*)(row + k0 + g8);
  u.q[1] = *(const uint4*)(row + k0 + 16 + g8);
  return u.v;
}
// swizzled k for computed A tiles (same mapping, element-wise form)
DEV int a_k_of(int lane, int h) {
  return ((h & 8) << 1) + ((lane >> 4) << 3) + (h & 7);
}
// B layout: lane L holds column n=L&15, half h -> k = (h>=8 ? 16 : 0) + (L>>4)*8 + (h&7), as A.
// Packed B-operand order: pk[((kk32*8 + nt)*32 + lane)*16 + h]  (contiguous).
// C/D layout: lane L, vgpr r -> m = (L>>4)*8 + r, n = L&15.

// ---- weight packing: row-major f32 (K x 128) -> WMMA-B packed bf16 ----------

__global__ void k_pack_b(const float* __restrict__ src, __bf16* __restrict__ dst,
                         int total) {
  int idx = blockIdx.x * 256 + threadIdx.x;          // over K*128
  if (idx >= total) return;
  int k = idx >> 7, n = idx & 127;
  int kl = k & 31, kk32 = k >> 5;
  int lane = (n & 15) | (((kl >> 3) & 1) << 4);
  int h = (kl & 7) | ((kl >> 4) << 3);
  int nt = n >> 4;
  dst[(((size_t)kk32 * 8 + nt) * 32 + lane) * 16 + h] = f2bf(src[idx]);
}

// ---- stage 1: embeddings + global LN ---------------------------------------

__global__ void k_embed_num(const float* __restrict__ inp, const float* __restrict__ nw,
                            const float* __restrict__ nb, float* __restrict__ feat,
                            float* __restrict__ stats) {
  int idx = blockIdx.x * 256 + threadIdx.x;          // over B*NUMC*H
  int b = idx / (NUMCc * Hd);
  int rem = idx - b * (NUMCc * Hd);
  int c = rem / Hd, h = rem - c * Hd;
  float x = inp[b * Cc + c];
  float v = fmaxf(x * nw[c * Hd + h] + nb[c * Hd + h], 0.f);
  feat[(b * Cc + c) * Hd + h] = v;
  float s = waveRedSum(v), q = waveRedSum(v * v);
  if ((threadIdx.x & 31) == 0) { atomicAdd(&stats[0], s); atomicAdd(&stats[1], q); }
}

__global__ void k_embed_cat(const float* __restrict__ inp, const float* __restrict__ ce,
                            float* __restrict__ feat, float* __restrict__ stats) {
  int idx = blockIdx.x * 256 + threadIdx.x;          // over B*CATC*H
  int b = idx / (CATCc * Hd);
  int rem = idx - b * (CATCc * Hd);
  int c = rem / Hd, h = rem - c * Hd;
  int ci = (int)inp[b * Cc + NUMCc + c];
  ci = min(max(ci, 0), CATVv - 1);
  float v = ce[((size_t)c * CATVv + ci) * Hd + h];
  feat[(b * Cc + NUMCc + c) * Hd + h] = v;
  float s = waveRedSum(v), q = waveRedSum(v * v);
  if ((threadIdx.x & 31) == 0) { atomicAdd(&stats[2], s); atomicAdd(&stats[3], q); }
}

__global__ void k_norm_feat(float* __restrict__ feat, __bf16* __restrict__ feat_bf,
                            const float* __restrict__ stats) {
  int idx = blockIdx.x * 256 + threadIdx.x;          // over B*C*H
  int c = (idx >> 7) & (Cc - 1);
  float mu, var;
  if (c < NUMCc) {
    float cnt = (float)(Bn * NUMCc * Hd);
    mu = stats[0] / cnt; var = stats[1] / cnt - mu * mu;
  } else {
    float cnt = (float)(Bn * CATCc * Hd);
    mu = stats[2] / cnt; var = stats[3] / cnt - mu * mu;
  }
  float v = (feat[idx] - mu) * rsqrtf(var + EPSf);
  feat[idx] = v;
  feat_bf[idx] = f2bf(v);
}

// ---- stage 2: h1 = LN(relu(feat@fw1+fb1)); imp = h1.fw2 + fb2 ---------------

__global__ __launch_bounds__(32)
void k_h1_imp(const __bf16* __restrict__ feat_bf, const __bf16* __restrict__ fw1pk,
              const float* __restrict__ fb1, const float* __restrict__ flng,
              const float* __restrict__ flnb, const float* __restrict__ fw2,
              const float* __restrict__ fb2, float* __restrict__ imp_raw) {
  int lane = threadIdx.x;
  int row0 = blockIdx.x * 16;                        // rows of flattened (B*C, H)
  const v16bf* Bp = (const v16bf*)fw1pk;
  v8f acc[8] = {};
  int m = lane & 15;
  const __bf16* arow = feat_bf + (size_t)(row0 + m) * Hd;
  for (int kk = 0; kk < Hd; kk += 32) {
    v16bf a = ld_a16(arow, kk, lane);
    int kk32 = kk >> 5;
    #pragma unroll
    for (int nt = 0; nt < 8; ++nt) {
      v16bf bt = Bp[((size_t)kk32 * 8 + nt) * 32 + lane];
      acc[nt] = wmma_bf16(a, bt, acc[nt]);
    }
  }
  int nlo = lane & 15, g = lane >> 4;
  #pragma unroll
  for (int r = 0; r < 8; ++r) {
    float vv[8]; float s = 0.f, q = 0.f;
    #pragma unroll
    for (int nt = 0; nt < 8; ++nt) {
      float v = fmaxf(acc[nt][r] + fb1[nt * 16 + nlo], 0.f);
      vv[nt] = v; s += v; q += v * v;
    }
    s = halfRedSum(s); q = halfRedSum(q);
    float mu = s * (1.f / Hd);
    float rs = rsqrtf(q * (1.f / Hd) - mu * mu + EPSf);
    float impp = 0.f;
    #pragma unroll
    for (int nt = 0; nt < 8; ++nt) {
      int nn = nt * 16 + nlo;
      float hn = (vv[nt] - mu) * rs * flng[nn] + flnb[nn];
      impp += hn * fw2[nn];
    }
    impp = halfRedSum(impp);
    if (nlo == 0) imp_raw[row0 + g * 8 + r] = impp + fb2[0];
  }
}

__global__ void k_imp_stats(const float* __restrict__ imp, float* __restrict__ statsI) {
  float s = 0.f, q = 0.f;
  for (int i = blockIdx.x * 256 + threadIdx.x; i < Bn * Cc; i += gridDim.x * 256) {
    float v = imp[i]; s += v; q += v * v;
  }
  s = waveRedSum(s); q = waveRedSum(q);
  if ((threadIdx.x & 31) == 0) { atomicAdd(&statsI[0], s); atomicAdd(&statsI[1], q); }
}

// ---- stage 3: LN(imp), top-K, masked softmax p, sorted selection ------------

__global__ __launch_bounds__(32)
void k_topk(const float* __restrict__ imp_raw, const float* __restrict__ statsI,
            float* __restrict__ imp_nrm, float* __restrict__ p,
            float* __restrict__ selF, int* __restrict__ selidx,
            int* __restrict__ selcnt) {
  int b = blockIdx.x, c = threadIdx.x;               // one wave per row, lane = column
  float cnt = (float)(Bn * Cc);
  float mu = statsI[0] / cnt;
  float rs = rsqrtf(statsI[1] / cnt - mu * mu + EPSf);
  float v = (imp_raw[b * Cc + c] - mu) * rs;
  imp_nrm[b * Cc + c] = v;
  int rank = 0;
  for (int c2 = 0; c2 < Cc; ++c2) {
    float v2 = __shfl(v, c2, 32);
    rank += (v2 > v) || (v2 == v && c2 < c);
  }
  bool sel = rank < Kk;
  float mv = sel ? v : -1e30f;
  #pragma unroll
  for (int o = 16; o; o >>= 1) mv = fmaxf(mv, __shfl_xor(mv, o, 32));
  float e = sel ? __expf(v - mv) : 0.f;
  float se = waveRedSum(e);
  p[b * Cc + c]    = e / se;
  selF[b * Cc + c] = sel ? 1.f : 0.f;
  unsigned bal = (unsigned)__ballot(sel);
  int slot = __popc(bal & ((1u << c) - 1u));
  if (sel) { selidx[b * Kk + slot] = c; atomicAdd(&selcnt[c], 1); }
}

__global__ void k_fe(const float* __restrict__ feat, const float* __restrict__ imp_nrm,
                     __bf16* __restrict__ fe_bf) {
  int idx = blockIdx.x * 256 + threadIdx.x;
  fe_bf[idx] = f2bf(feat[idx] * imp_nrm[idx >> 7]);
}

// ---- stage 4: h_proj = fe @ gcn_w (1024x4096x128); also pack h_proj as B ----

__global__ __launch_bounds__(32)
void k_hproj(const __bf16* __restrict__ fe_bf, const __bf16* __restrict__ gwpk,
             float* __restrict__ h_proj, __bf16* __restrict__ hppk) {
  int lane = threadIdx.x;
  int row0 = blockIdx.x * 16;
  const v16bf* Bp = (const v16bf*)gwpk;
  v8f acc[8] = {};
  int m = lane & 15;
  const __bf16* arow = fe_bf + (size_t)(row0 + m) * (Cc * Hd);
  for (int kk = 0; kk < Cc * Hd; kk += 32) {
    v16bf a = ld_a16(arow, kk, lane);
    int kk32 = kk >> 5;
    #pragma unroll
    for (int nt = 0; nt < 8; ++nt) {
      v16bf bt = Bp[((size_t)kk32 * 8 + nt) * 32 + lane];
      acc[nt] = wmma_bf16(a, bt, acc[nt]);
    }
  }
  int nlo = lane & 15, g = lane >> 4;
  #pragma unroll
  for (int nt = 0; nt < 8; ++nt) {
    #pragma unroll
    for (int r = 0; r < 8; ++r) {
      int j = row0 + g * 8 + r;                      // output row = GCN k index
      float v = acc[nt][r];
      h_proj[(size_t)j * Hd + nt * 16 + nlo] = v;
      int kl = j & 31, kk32 = j >> 5;                // pack as B operand for GCN
      int plane = nlo | (((kl >> 3) & 1) << 4);
      hppk[(((size_t)kk32 * 8 + nt) * 32 + plane) * 16 + ((kl & 7) | ((kl >> 4) << 3))] = f2bf(v);
    }
  }
}

// ---- stage 5: GCN attention (streaming; a = impt@impt.T never materialized) -
// impt[b][c] = p[b][c] * sel[b][t] * (c != t)

__global__ __launch_bounds__(256)
void k_amax(const float* __restrict__ p, const float* __restrict__ selF,
            unsigned* __restrict__ amaxU) {
  int t = blockIdx.y;
  int j = blockIdx.x * 256 + threadIdx.x;
  __shared__ float tile[256 * Cc];
  float rj[Cc];
  float rmask = selF[j * Cc + t];
  #pragma unroll
  for (int c = 0; c < Cc; ++c)
    rj[c] = p[j * Cc + c] * rmask * ((c != t) ? 1.f : 0.f);
  float lmax = 0.f;
  for (int i0 = 0; i0 < Bn; i0 += 256) {
    __syncthreads();
    for (int idx = threadIdx.x; idx < 256 * Cc; idx += 256) {
      int r = idx >> 5, c = idx & 31; int i = i0 + r;
      tile[idx] = p[i * Cc + c] * selF[i * Cc + t] * ((c != t) ? 1.f : 0.f);
    }
    __syncthreads();
    for (int r = 0; r < 256; ++r) {
      if (i0 + r == j) continue;
      float d = 0.f;
      #pragma unroll
      for (int c = 0; c < Cc; ++c) d += rj[c] * tile[r * Cc + c];
      lmax = fmaxf(lmax, d);
    }
  }
  if (lmax > 0.f) atomicMax(&amaxU[t], __float_as_uint(lmax));  // nonneg floats
}

__global__ __launch_bounds__(256)
void k_colsum(const float* __restrict__ p, const float* __restrict__ selF,
              const float* __restrict__ amax, float* __restrict__ colsum,
              float* __restrict__ Sw) {
  int t = blockIdx.y;
  int j = blockIdx.x * 256 + threadIdx.x;
  __shared__ float tile[256 * Cc];
  float rj[Cc];
  float rmask = selF[j * Cc + t];
  #pragma unroll
  for (int c = 0; c < Cc; ++c)
    rj[c] = p[j * Cc + c] * rmask * ((c != t) ? 1.f : 0.f);
  float am = amax[t];
  float cs = 0.f;
  for (int i0 = 0; i0 < Bn; i0 += 256) {
    __syncthreads();
    for (int idx = threadIdx.x; idx < 256 * Cc; idx += 256) {
      int r = idx >> 5, c = idx & 31; int i = i0 + r;
      tile[idx] = p[i * Cc + c] * selF[i * Cc + t] * ((c != t) ? 1.f : 0.f);
    }
    __syncthreads();
    for (int r = 0; r < 256; ++r) {
      if (i0 + r == j) continue;
      float d = 0.f;
      #pragma unroll
      for (int c = 0; c < Cc; ++c) d += rj[c] * tile[r * Cc + c];
      if (d > 0.f) cs += __expf(d - am);
    }
  }
  colsum[t * Bn + j] = cs;            // a is symmetric: colsum == rowsum
  atomicAdd(&Sw[t], cs);
}

__global__ void k_dinv(const float* __restrict__ selF, const float* __restrict__ colsum,
                       const float* __restrict__ Sw, float* __restrict__ dinv) {
  int idx = blockIdx.x * 256 + threadIdx.x;          // t*1024 + j
  int t = idx >> 10, j = idx & (Bn - 1);
  float m  = selF[j * Cc + t];
  float Sv = Sw[t];
  float ws = (Sv > 0.f) ? colsum[idx] / Sv : 0.f;
  dinv[idx] = (m > 0.f) ? rsqrtf(ws + m) : 0.f;      // deg = colsum(w)/S + m
}

// out_i = relu( dinv_i/S * sum_j (w_ij dinv_j) hp_j + dinv_i^2 m_i hp_i + b )
// dinv_j folded into A -> B is the t-independent pre-packed h_proj.
__global__ __launch_bounds__(32)
void k_gcn_gemm(const float* __restrict__ p, const float* __restrict__ selF,
                const float* __restrict__ dinv, const float* __restrict__ h_proj,
                const __bf16* __restrict__ hppk, const float* __restrict__ gcn_b,
                const float* __restrict__ amax, const float* __restrict__ Sw,
                float* __restrict__ out_all, float* __restrict__ msum,
                float* __restrict__ msumsq) {
  int lane = threadIdx.x;
  int t  = blockIdx.y;
  int i0 = blockIdx.x * 16;
  __shared__ float ldsI[16 * Cc];
  __shared__ float ldsJ[32 * Cc];
  __shared__ float ldsDJ[32];
  __shared__ float dinvI[16], mI[16];
  for (int idx = lane; idx < 16 * Cc; idx += 32) {
    int r = idx >> 5, c = idx & 31;
    ldsI[idx] = p[(i0 + r) * Cc + c] * selF[(i0 + r) * Cc + t] * ((c != t) ? 1.f : 0.f);
  }
  if (lane < 16) {
    dinvI[lane] = dinv[t * Bn + i0 + lane];
    mI[lane]    = selF[(i0 + lane) * Cc + t];
  }
  __syncthreads();
  float am = amax[t];
  float Sv = Sw[t];
  float Sinv = (Sv > 0.f) ? 1.f / Sv : 0.f;
  const v16bf* Bp = (const v16bf*)hppk;
  v8f acc[8] = {};
  int m = lane & 15;
  for (int j0 = 0; j0 < Bn; j0 += 32) {
    __syncthreads();
    for (int idx = lane; idx < 32 * Cc; idx += 32) {
      int r = idx >> 5, c = idx & 31; int j = j0 + r;
      ldsJ[idx] = p[j * Cc + c] * selF[j * Cc + t] * ((c != t) ? 1.f : 0.f);
    }
    ldsDJ[lane] = dinv[t * Bn + j0 + lane];
    __syncthreads();
    v16bf a;
    #pragma unroll
    for (int h = 0; h < 16; ++h) {
      int jl = a_k_of(lane, h);
      float w = 0.f;
      if (i0 + m != j0 + jl) {
        float d = 0.f;
        #pragma unroll
        for (int c = 0; c < Cc; ++c) d += ldsI[m * Cc + c] * ldsJ[jl * Cc + c];
        w = (d > 0.f) ? __expf(d - am) * ldsDJ[jl] : 0.f;
      }
      a[h] = f2bf(w);
    }
    int kk32 = j0 >> 5;
    #pragma unroll
    for (int nt = 0; nt < 8; ++nt) {
      v16bf bt = Bp[((size_t)kk32 * 8 + nt) * 32 + lane];
      acc[nt] = wmma_bf16(a, bt, acc[nt]);
    }
  }
  int nlo = lane & 15, g = lane >> 4;
  float ls = 0.f, lq = 0.f;
  #pragma unroll
  for (int nt = 0; nt < 8; ++nt) {
    #pragma unroll
    for (int r = 0; r < 8; ++r) {
      int mr = g * 8 + r; int i = i0 + mr; int nn = nt * 16 + nlo;
      float val = dinvI[mr] * Sinv * acc[nt][r]
                + dinvI[mr] * dinvI[mr] * mI[mr] * h_proj[(size_t)i * Hd + nn]
                + gcn_b[nn];
      val = fmaxf(val, 0.f);
      out_all[((size_t)t * Bn + i) * Hd + nn] = val;
      if (mI[mr] > 0.f) { ls += val; lq += val * val; }
    }
  }
  ls = waveRedSum(ls); lq = waveRedSum(lq);
  if (lane == 0) { atomicAdd(&msum[t], ls); atomicAdd(&msumsq[t], lq); }
}

__global__ void k_gcn_norm(const float* __restrict__ out_all,
                           __bf16* __restrict__ out_all_bf,
                           const float* __restrict__ msum,
                           const float* __restrict__ msumsq,
                           const int* __restrict__ selcnt) {
  size_t idx = (size_t)blockIdx.x * 256 + threadIdx.x;   // over C*B*H
  int t = (int)(idx >> 17);                              // 1024*128 = 2^17
  float cnt = (float)selcnt[t] * (float)Hd;
  float v = out_all[idx];
  if (cnt > 0.f) {
    float mu = msum[t] / cnt;
    float rs = rsqrtf(msumsq[t] / cnt - mu * mu + EPSf);
    v = (v - mu) * rs;
  }
  out_all_bf[idx] = f2bf(v);
}

// ---- stage 6: pd = [processed | fe]; h2 = LN(relu(pd@pw1+pb1)); out = h2@pw2 -

__global__ __launch_bounds__(32)
void k_final(const __bf16* __restrict__ out_all_bf, const __bf16* __restrict__ fe_bf,
             const int* __restrict__ selidx, const __bf16* __restrict__ pw1pk,
             const float* __restrict__ pb1, const float* __restrict__ plng,
             const float* __restrict__ plnb, const float* __restrict__ pw2,
             const float* __restrict__ pb2, float* __restrict__ out) {
  __shared__ int ldsSel[16 * Kk];
  int lane = threadIdx.x;
  int row0 = blockIdx.x * 16;
  for (int idx = lane; idx < 16 * Kk; idx += 32)
    ldsSel[idx] = selidx[(row0 + (idx >> 4)) * Kk + (idx & 15)];
  __syncthreads();
  const v16bf* Bp = (const v16bf*)pw1pk;
  v8f acc[8] = {};
  int m = lane & 15;
  int brow = row0 + m;
  int g8 = (lane >> 4) << 3;
  for (int q0 = 0; q0 < (Kk + Cc) * Hd; q0 += 32) {
    // two contiguous 8-element runs; neither straddles a 128-block or the
    // processed/fe boundary (q0 is a multiple of 32).
    union { uint4 q[2]; v16bf v; } u;
    #pragma unroll
    for (int run = 0; run < 2; ++run) {
      int q = q0 + run * 16 + g8;
      const __bf16* ptr;
      if (q < Kk * Hd) {
        int t = ldsSel[m * Kk + (q >> 7)];
        ptr = out_all_bf + ((size_t)t * Bn + brow) * Hd + (q & 127);
      } else {
        ptr = fe_bf + (size_t)brow * (Cc * Hd) + (q - Kk * Hd);
      }
      u.q[run] = *(const uint4*)ptr;
    }
    v16bf a = u.v;
    int kk32 = q0 >> 5;
    #pragma unroll
    for (int nt = 0; nt < 8; ++nt) {
      v16bf bt = Bp[((size_t)kk32 * 8 + nt) * 32 + lane];
      acc[nt] = wmma_bf16(a, bt, acc[nt]);
    }
  }
  int nlo = lane & 15, g = lane >> 4;
  #pragma unroll
  for (int r = 0; r < 8; ++r) {
    float vv[8]; float s = 0.f, q = 0.f;
    #pragma unroll
    for (int nt = 0; nt < 8; ++nt) {
      float v = fmaxf(acc[nt][r] + pb1[nt * 16 + nlo], 0.f);
      vv[nt] = v; s += v; q += v * v;
    }
    s = halfRedSum(s); q = halfRedSum(q);
    float mu = s * (1.f / Hd);
    float rs = rsqrtf(q * (1.f / Hd) - mu * mu + EPSf);
    float o0 = 0.f, o1 = 0.f;
    #pragma unroll
    for (int nt = 0; nt < 8; ++nt) {
      int nn = nt * 16 + nlo;
      float hn = (vv[nt] - mu) * rs * plng[nn] + plnb[nn];
      o0 += hn * pw2[nn * 2 + 0];
      o1 += hn * pw2[nn * 2 + 1];
    }
    o0 = halfRedSum(o0); o1 = halfRedSum(o1);
    if (nlo == 0) {
      int rr = row0 + g * 8 + r;
      out[rr * 2 + 0] = o0 + pb2[0];
      out[rr * 2 + 1] = o1 + pb2[1];
    }
  }
}

// ---------------------------------------------------------------------------

extern "C" void kernel_launch(void* const* d_in, const int* in_sizes, int n_in,
                              void* d_out, int out_size, void* d_ws, size_t ws_size,
                              hipStream_t stream) {
  (void)in_sizes; (void)n_in; (void)out_size; (void)ws_size;
  const float* inp     = (const float*)d_in[0];
  const float* num_w   = (const float*)d_in[1];
  const float* num_b   = (const float*)d_in[2];
  const float* cat_emb = (const float*)d_in[3];
  const float* fw1     = (const float*)d_in[4];
  const float* fb1     = (const float*)d_in[5];
  const float* fln_g   = (const float*)d_in[6];
  const float* fln_b   = (const float*)d_in[7];
  const float* fw2     = (const float*)d_in[8];
  const float* fb2     = (const float*)d_in[9];
  const float* gcn_w   = (const float*)d_in[10];
  const float* gcn_b   = (const float*)d_in[11];
  const float* pw1     = (const float*)d_in[12];
  const float* pb1     = (const float*)d_in[13];
  const float* pln_g   = (const float*)d_in[14];
  const float* pln_b   = (const float*)d_in[15];
  const float* pw2     = (const float*)d_in[16];
  const float* pb2     = (const float*)d_in[17];

  // workspace carve-up (~63 MB); every offset is a multiple of 8 floats (32B)
  float* base = (float*)d_ws;
  size_t off = 0;
  float* feat     = base + off; off += (size_t)Bn * Cc * Hd;       // f32
  float* out_all  = base + off; off += (size_t)Cc * Bn * Hd;       // f32
  float* h_proj   = base + off; off += (size_t)Bn * Hd;
  float* imp_raw  = base + off; off += Bn * Cc;
  float* imp_nrm  = base + off; off += Bn * Cc;
  float* p        = base + off; off += Bn * Cc;
  float* selF     = base + off; off += Bn * Cc;
  float* dinv     = base + off; off += (size_t)Cc * Bn;
  int*   selidx   = (int*)(base + off); off += Bn * Kk;
  __bf16* feat_bf = (__bf16*)(base + off); off += (size_t)Bn * Cc * Hd / 2;
  __bf16* fe_bf   = (__bf16*)(base + off); off += (size_t)Bn * Cc * Hd / 2;
  __bf16* oab     = (__bf16*)(base + off); off += (size_t)Cc * Bn * Hd / 2;
  __bf16* hppk    = (__bf16*)(base + off); off += (size_t)Bn * Hd / 2;
  __bf16* fw1pk   = (__bf16*)(base + off); off += (Hd * Hd) / 2;
  __bf16* gwpk    = (__bf16*)(base + off); off += (Cc * Hd * Hd) / 2;
  __bf16* pw1pk   = (__bf16*)(base + off); off += ((Kk + Cc) * Hd * Hd) / 2;
  float* zbase    = base + off;
  float*    stats  = zbase;                  // 4: num sum/sq, cat sum/sq
  float*    statsI = zbase + 4;              // 2: imp sum/sq
  unsigned* amaxU  = (unsigned*)(zbase + 6); // 32
  float*    Sw     = zbase + 38;             // 32
  float*    colsum = zbase + 70;             // 32*1024
  float*    msum   = colsum + Cc * Bn;       // 32
  float*    msumsq = msum + Cc;              // 32
  int*      selcnt = (int*)(msumsq + Cc);    // 32
  size_t zbytes = (size_t)(4 + 2 + 32 + 32 + Cc * Bn + 32 + 32 + 32) * sizeof(float);

  hipMemsetAsync(zbase, 0, zbytes, stream);

  int nfw1 = Hd * Hd, ngcn = Cc * Hd * Hd, npw1 = (Kk + Cc) * Hd * Hd;
  k_pack_b<<<(nfw1 + 255) / 256, 256, 0, stream>>>(fw1,   fw1pk, nfw1);
  k_pack_b<<<(ngcn + 255) / 256, 256, 0, stream>>>(gcn_w, gwpk,  ngcn);
  k_pack_b<<<(npw1 + 255) / 256, 256, 0, stream>>>(pw1,   pw1pk, npw1);

  k_embed_num<<<(Bn * NUMCc * Hd) / 256, 256, 0, stream>>>(inp, num_w, num_b, feat, stats);
  k_embed_cat<<<(Bn * CATCc * Hd) / 256, 256, 0, stream>>>(inp, cat_emb, feat, stats);
  k_norm_feat<<<(Bn * Cc * Hd) / 256, 256, 0, stream>>>(feat, feat_bf, stats);

  k_h1_imp<<<(Bn * Cc) / 16, 32, 0, stream>>>(feat_bf, fw1pk, fb1, fln_g, fln_b,
                                              fw2, fb2, imp_raw);
  k_imp_stats<<<64, 256, 0, stream>>>(imp_raw, statsI);
  k_topk<<<Bn, 32, 0, stream>>>(imp_raw, statsI, imp_nrm, p, selF, selidx, selcnt);
  k_fe<<<(Bn * Cc * Hd) / 256, 256, 0, stream>>>(feat, imp_nrm, fe_bf);

  k_hproj<<<Bn / 16, 32, 0, stream>>>(fe_bf, gwpk, h_proj, hppk);

  k_amax  <<<dim3(Bn / 256, Cc), 256, 0, stream>>>(p, selF, amaxU);
  k_colsum<<<dim3(Bn / 256, Cc), 256, 0, stream>>>(p, selF, (const float*)amaxU,
                                                   colsum, Sw);
  k_dinv  <<<(Cc * Bn) / 256, 256, 0, stream>>>(selF, colsum, Sw, dinv);
  k_gcn_gemm<<<dim3(Bn / 16, Cc), 32, 0, stream>>>(p, selF, dinv, h_proj, hppk,
                                                   gcn_b, (const float*)amaxU, Sw,
                                                   out_all, msum, msumsq);
  k_gcn_norm<<<(Cc * Bn * Hd) / 256, 256, 0, stream>>>(out_all, oab, msum, msumsq,
                                                       selcnt);

  k_final<<<Bn / 16, 32, 0, stream>>>(oab, fe_bf, selidx, pw1pk, pb1, pln_g, pln_b,
                                      pw2, pb2, (float*)d_out);
}
